// CGCNN_7172595384787
// MI455X (gfx1250) — compile-verified, matrix operand order checked by reading, not run on hardware
//
#include <hip/hip_runtime.h>
#include <hip/hip_bf16.h>
#include <math.h>

// ---------------- problem constants (match reference) ----------------
#define N_NODES   50000
#define ATOM_DIM  92
#define EDGE_DIM  41
#define N_LAYERS  4
#define HIDDEN    128
#define N_GRAPHS  256
#define BN_EPS    1e-5f
#define NEG_SLOPE 0.01f

#define HPAD   96      // ATOM_DIM padded to 6*16
#define NTILES 6       // 96 / 16
#define KTILES 8       // K padded: 96(dst)+96(src)+64(edge) = 256 = 8*32
// fragment storage: per frag 32 lanes x 16 bf16 = 512 halves (1 KB)
#define FRAG_HALVES        (32 * 16)
#define FRAGS_PER_LAYER    (2 * KTILES * NTILES)         // 96
#define FRAG_HALVES_LAYER  (FRAGS_PER_LAYER * FRAG_HALVES)

typedef __attribute__((ext_vector_type(16))) __bf16 v16bf;
typedef __attribute__((ext_vector_type(8)))  float  v8f;

union Frag {
    uint4  u[2];
    v16bf  v;
    __bf16 h[16];
};

__device__ __forceinline__ v8f wmma_bf16(v16bf a, v16bf b, v8f c) {
    // D = A(16x32 bf16) * B(32x16 bf16) + C(16x16 f32)
    return __builtin_amdgcn_wmma_f32_16x16x32_bf16(
        /*neg_a=*/false, a, /*neg_b=*/false, b,
        /*c_mod=*/(short)0, c, /*reuse_a=*/false, /*reuse_b=*/false);
}

// fast transcendentals: v_exp_f32 / v_log_f32 / v_rcp_f32 hardware paths
__device__ __forceinline__ float dev_sigmoid(float x) {
    return __builtin_amdgcn_rcpf(1.0f + __expf(-x));
}
__device__ __forceinline__ float dev_softplus(float x) {
    // stable: max(x,0) + log(1 + exp(-|x|))
    float e = __expf(-fabsf(x));
    return fmaxf(x, 0.0f) + __logf(1.0f + e);
}
__device__ __forceinline__ float dev_leaky(float x) { return x > 0.0f ? x : NEG_SLOPE * x; }

// ---------------- weight fragment pre-swizzle ----------------
// B-fragment layout assumption (wave32, 16-bit B 32x16): lane n (and lane n+16)
// hold column n; half-lane h contributes K = k0 + h*8 + i (i<8) and
// K = k0 + 16 + h*8 + (i-8) (i>=8) -- the mirror of the documented A layout.
__global__ void prep_frags(const float* __restrict__ Wf,
                           const float* __restrict__ Ws,
                           __bf16* __restrict__ frag) {
    int f    = blockIdx.x;       // 0 .. 4*2*8*6-1 = 383
    int lane = threadIdx.x;      // 0..31
    int n  = f % NTILES;  int t = f / NTILES;
    int kt = t % KTILES;  t /= KTILES;
    int w  = t % 2;       int L = t / 2;
    const float* W = (w == 0 ? Wf : Ws) + (size_t)L * (2 * ATOM_DIM + EDGE_DIM) * ATOM_DIM;
    int col  = n * 16 + (lane & 15);
    int half = lane >> 4;
    __bf16* out = frag + ((size_t)f * 32 + lane) * 16;
    #pragma unroll
    for (int i = 0; i < 16; ++i) {
        int kp = kt * 32 + half * 8 + i + (i >= 8 ? 8 : 0);  // padded K index 0..255
        int zr;                                               // row in original 225-row W
        if (kp < HPAD)            { zr = (kp < ATOM_DIM) ? kp : -1; }
        else if (kp < 2 * HPAD)   { int r = kp - HPAD;     zr = (r < ATOM_DIM) ? ATOM_DIM + r : -1; }
        else                      { int r = kp - 2 * HPAD; zr = (r < EDGE_DIM) ? 2 * ATOM_DIM + r : -1; }
        float v = (zr >= 0 && col < ATOM_DIM) ? W[(size_t)zr * ATOM_DIM + col] : 0.0f;
        out[i] = (__bf16)v;
    }
}

// ---------------- init / utility ----------------
__global__ void init_h(const float* __restrict__ x,
                       float* __restrict__ hcur, __bf16* __restrict__ hbf) {
    int idx = blockIdx.x * blockDim.x + threadIdx.x;   // over N_NODES*HPAD
    if (idx >= N_NODES * HPAD) return;
    int r = idx / HPAD, c = idx % HPAD;
    if (c < ATOM_DIM) {
        float v = x[(size_t)r * ATOM_DIM + c];
        hcur[(size_t)r * ATOM_DIM + c] = v;
        hbf[idx] = (__bf16)v;
    } else {
        hbf[idx] = (__bf16)0.0f;                        // zero padding, never rewritten
    }
}

__global__ void zero_f32(float* __restrict__ p, int n) {
    int i = blockIdx.x * blockDim.x + threadIdx.x;
    if (i < n) p[i] = 0.0f;
}

// ---------------- the WMMA edge-message kernel ----------------
// One wave per 16-edge tile. Per k-step: 1 A-fragment + 12 B-fragments loaded in
// one burst, then 12 back-to-back v_wmma_f32_16x16x32_bf16 (96 total per tile).
__global__ void __launch_bounds__(256)
edge_gemm_scatter(const __bf16* __restrict__ hbf,     // [N, 96] bf16 (padded)
                  const float*  __restrict__ eattr,   // [E, 41] f32
                  const int*    __restrict__ srcIdx,
                  const int*    __restrict__ dstIdx,
                  const __bf16* __restrict__ frag,    // this layer's 96 B-fragments
                  const float*  __restrict__ biasF,   // [92]
                  const float*  __restrict__ biasS,   // [92]
                  float*        __restrict__ agg,     // [N, 92] f32 (atomic target)
                  int numTiles)
{
    const int lane = threadIdx.x & 31;
    const int wv   = threadIdx.x >> 5;
    const int tile = blockIdx.x * 8 + wv;
    if (tile >= numTiles) return;          // wave-uniform -> EXEC stays all-ones below
    const int e0   = tile * 16;
    const int m    = lane & 15;
    const int half = lane >> 4;
    const long eM  = (long)e0 + m;
    const int d = dstIdx[eM];
    const int s = srcIdx[eM];

    // fold per-column bias into accumulator init: each lane's 8 C rows share
    // one column (N = lane&15 within the n-tile), so splat bias across rows.
    v8f accF[NTILES];
    v8f accS[NTILES];
    #pragma unroll
    for (int n = 0; n < NTILES; ++n) {
        const int col = n * 16 + (lane & 15);
        const float bFn = (col < ATOM_DIM) ? biasF[col] : 0.0f;
        const float bSn = (col < ATOM_DIM) ? biasS[col] : 0.0f;
        v8f tF, tS;
        #pragma unroll
        for (int r = 0; r < 8; ++r) { tF[r] = bFn; tS[r] = bSn; }
        accF[n] = tF; accS[n] = tS;
    }

    const uint4* h4 = (const uint4*)hbf;   // 8 bf16 per uint4
    const uint4* w4 = (const uint4*)frag;

    #pragma unroll
    for (int ktg = 0; ktg < KTILES; ++ktg) {
        // ---- A fragment (documented 16-bit A 16x32 layout) ----
        Frag A;
        if (ktg < 6) {
            // dst (ktg 0..2) / src (ktg 3..5): contiguous 16B loads from bf16 table
            const int node = (ktg < 3) ? d : s;
            const int kt   = (ktg < 3) ? ktg : ktg - 3;
            const long b = ((long)node * HPAD + kt * 32 + half * 8) >> 3; // uint4 units
            A.u[0] = h4[b];
            A.u[1] = h4[b + 2];                 // +16 halves
        } else {
            // edge_attr (ktg 6..7): fp32 -> bf16 on the fly, zero-pad K to 64
            const int kb = (ktg - 6) * 32 + half * 8;
            #pragma unroll
            for (int i = 0; i < 16; ++i) {
                const int K = kb + i + (i >= 8 ? 8 : 0);
                float v = (K < EDGE_DIM) ? eattr[eM * EDGE_DIM + K] : 0.0f;
                A.h[i] = (__bf16)v;
            }
        }
        // ---- all 12 B fragments for this k-step in one burst ----
        Frag B[2 * NTILES];
        #pragma unroll
        for (int q = 0; q < 2 * NTILES; ++q) {
            const int w = (q >= NTILES) ? 1 : 0;
            const int n = (q >= NTILES) ? q - NTILES : q;
            const long o = ((long)((w * KTILES + ktg) * NTILES + n) * 32 + lane) * 2;
            B[q].u[0] = w4[o];
            B[q].u[1] = w4[o + 1];
        }
        // ---- 12 back-to-back WMMAs ----
        #pragma unroll
        for (int n = 0; n < NTILES; ++n) accF[n] = wmma_bf16(A.v, B[n].v, accF[n]);
        #pragma unroll
        for (int n = 0; n < NTILES; ++n) accS[n] = wmma_bf16(A.v, B[NTILES + n].v, accS[n]);
    }

    // C/D layout: VGPR r holds row M = half*8 + r, column N = lane&15 (per n-tile)
    int dRow[8];
    #pragma unroll
    for (int r = 0; r < 8; ++r) dRow[r] = dstIdx[e0 + half * 8 + r];

    #pragma unroll
    for (int n = 0; n < NTILES; ++n) {
        const int col = n * 16 + (lane & 15);
        if (col < ATOM_DIM) {
            #pragma unroll
            for (int r = 0; r < 8; ++r) {
                float g  = dev_sigmoid(accF[n][r]);
                float sp = dev_softplus(accS[n][r]);
                atomicAdd(&agg[(long)dRow[r] * ATOM_DIM + col], g * sp);
            }
        }
    }
}

// ---------------- batchnorm (training-mode batch stats, biased var) ----------------
__global__ void bn_stats(const float* __restrict__ agg, float* __restrict__ stats) {
    const int c = blockIdx.x;                       // channel 0..91
    float s = 0.0f, s2 = 0.0f;
    for (int r = threadIdx.x; r < N_NODES; r += blockDim.x) {
        float v = agg[(long)r * ATOM_DIM + c];
        s += v; s2 += v * v;
    }
    __shared__ float rs[256], rq[256];
    rs[threadIdx.x] = s; rq[threadIdx.x] = s2; __syncthreads();
    for (int o = 128; o > 0; o >>= 1) {
        if (threadIdx.x < o) { rs[threadIdx.x] += rs[threadIdx.x + o]; rq[threadIdx.x] += rq[threadIdx.x + o]; }
        __syncthreads();
    }
    if (threadIdx.x == 0) {
        float mean = rs[0] / (float)N_NODES;
        float var  = rq[0] / (float)N_NODES - mean * mean;
        stats[c] = mean; stats[ATOM_DIM + c] = var;
    }
}

__global__ void bn_apply(float* __restrict__ agg, float* __restrict__ hcur,
                         __bf16* __restrict__ hbf, const float* __restrict__ stats,
                         const float* __restrict__ gamma, const float* __restrict__ beta) {
    long idx = (long)blockIdx.x * blockDim.x + threadIdx.x;   // over N*ATOM_DIM
    if (idx >= (long)N_NODES * ATOM_DIM) return;
    int  c = (int)(idx % ATOM_DIM);
    long r = idx / ATOM_DIM;
    float mean = stats[c], var = stats[ATOM_DIM + c];
    float v = (agg[idx] - mean) * rsqrtf(var + BN_EPS) * gamma[c] + beta[c] + hcur[idx];
    v = dev_leaky(v);
    hcur[idx] = v;
    hbf[r * HPAD + c] = (__bf16)v;
    agg[idx] = 0.0f;                                 // ready for next layer's scatter
}

// ---------------- final MLP + scatter-mean pooling ----------------
__global__ void __launch_bounds__(HIDDEN)
mlp_pool(const float* __restrict__ hcur, const int* __restrict__ batch,
         const float* __restrict__ W1, const float* __restrict__ b1,
         const float* __restrict__ W2, const float* __restrict__ b2,
         float* __restrict__ sums, float* __restrict__ cnt) {
    const int v = blockIdx.x;        // node
    const int j = threadIdx.x;       // hidden unit
    const float* hv = hcur + (long)v * ATOM_DIM;
    float acc = b1[j];
    #pragma unroll 4
    for (int k = 0; k < ATOM_DIM; ++k) acc += hv[k] * W1[k * HIDDEN + j];
    acc = dev_leaky(acc);
    float contrib = acc * W2[j];
    __shared__ float red[HIDDEN];
    red[j] = contrib; __syncthreads();
    for (int o = HIDDEN / 2; o > 0; o >>= 1) {
        if (j < o) red[j] += red[j + o];
        __syncthreads();
    }
    if (j == 0) {
        int g = batch[v];
        atomicAdd(&sums[g], red[0] + b2[0]);
        atomicAdd(&cnt[g], 1.0f);
    }
}

__global__ void finalize_pool(const float* __restrict__ sums, const float* __restrict__ cnt,
                              float* __restrict__ out) {
    int g = blockIdx.x * blockDim.x + threadIdx.x;
    if (g < N_GRAPHS) out[g] = sums[g] / fmaxf(cnt[g], 1.0f);
}

// ---------------- host-side launch ----------------
extern "C" void kernel_launch(void* const* d_in, const int* in_sizes, int n_in,
                              void* d_out, int out_size, void* d_ws, size_t ws_size,
                              hipStream_t stream) {
    const float* x     = (const float*)d_in[0];
    const int*   ei    = (const int*)  d_in[1];
    const float* eattr = (const float*)d_in[2];
    const int*   batch = (const int*)  d_in[3];
    const float* Wf    = (const float*)d_in[4];
    const float* bf    = (const float*)d_in[5];
    const float* Ws    = (const float*)d_in[6];
    const float* bs    = (const float*)d_in[7];
    const float* gamma = (const float*)d_in[8];
    const float* beta  = (const float*)d_in[9];
    const float* W1    = (const float*)d_in[10];
    const float* b1    = (const float*)d_in[11];
    const float* W2    = (const float*)d_in[12];
    const float* b2    = (const float*)d_in[13];

    const int E = in_sizes[2] / EDGE_DIM;            // edge_attr is [E, 41]
    const int* srcIdx = ei;                          // edge_index[0]
    const int* dstIdx = ei + E;                      // edge_index[1]

    // workspace carve-up (~47 MB total)
    char* p = (char*)d_ws;
    auto alloc = [&](size_t bytes) -> char* {
        char* r = p; p += (bytes + 255) & ~(size_t)255; return r;
    };
    float*  hcur  = (float*) alloc((size_t)N_NODES * ATOM_DIM * 4);
    float*  agg   = (float*) alloc((size_t)N_NODES * ATOM_DIM * 4);
    __bf16* hbf   = (__bf16*)alloc((size_t)N_NODES * HPAD * 2);
    __bf16* frag  = (__bf16*)alloc((size_t)N_LAYERS * FRAG_HALVES_LAYER * 2);
    float*  stats = (float*) alloc(2 * ATOM_DIM * 4);
    float*  sums  = (float*) alloc(N_GRAPHS * 4);
    float*  cnt   = (float*) alloc(N_GRAPHS * 4);

    // 1) pre-swizzle all layer weights into WMMA B-fragments
    prep_frags<<<N_LAYERS * FRAGS_PER_LAYER, 32, 0, stream>>>(Wf, Ws, frag);

    // 2) init node tables + zero accumulators (harness does not re-poison between replays)
    init_h<<<(N_NODES * HPAD + 255) / 256, 256, 0, stream>>>(x, hcur, hbf);
    zero_f32<<<(N_NODES * ATOM_DIM + 255) / 256, 256, 0, stream>>>(agg, N_NODES * ATOM_DIM);
    zero_f32<<<1, 256, 0, stream>>>(sums, N_GRAPHS);
    zero_f32<<<1, 256, 0, stream>>>(cnt,  N_GRAPHS);

    // 3) CGConv layers
    const int numTiles = E / 16;                     // E = 1.6M divides exactly
    const int blocks   = (numTiles + 7) / 8;         // 8 waves (tiles) per 256-thread block
    for (int L = 0; L < N_LAYERS; ++L) {
        edge_gemm_scatter<<<blocks, 256, 0, stream>>>(
            hbf, eattr, srcIdx, dstIdx,
            frag + (size_t)L * FRAG_HALVES_LAYER,
            bf + L * ATOM_DIM, bs + L * ATOM_DIM,
            agg, numTiles);
        bn_stats<<<ATOM_DIM, 256, 0, stream>>>(agg, stats);
        bn_apply<<<(N_NODES * ATOM_DIM + 255) / 256, 256, 0, stream>>>(
            agg, hcur, hbf, stats, gamma + L * ATOM_DIM, beta + L * ATOM_DIM);
    }

    // 4) MLP head + scatter-mean pooling
    mlp_pool<<<N_NODES, HIDDEN, 0, stream>>>(hcur, batch, W1, b1, W2, b2, sums, cnt);
    finalize_pool<<<1, 256, 0, stream>>>(sums, cnt, (float*)d_out);
}
